// SelfAttention_17248588661477
// MI455X (gfx1250) — matrix operand order repaired and checked by reading, not compiled, not run on hardware
//
#include <hip/hip_runtime.h>
#include <hip/hip_bf16.h>

// ---------------------------------------------------------------------------
// Fused self-attention layer for MI455X (gfx1250, wave32, WMMA + TDM)
//   B=2, T=2048, HID=1024, NH=16, HD=64
// Pipeline:
//   1) qkv_rope_gemm : q/k/v = x @ W^T  (f16 WMMA, f32 acc) + RoPE on q,k
//   2) flash_attn    : online-softmax causal attention; Q/K/V tiles pulled
//                      into LDS by the Tensor Data Mover (tensor_load_to_lds)
//   3) out_proj_gemm : out = attn @ Wo^T  (f32 result)
// ---------------------------------------------------------------------------

#define BB   2
#define TT   2048
#define HID  1024
#define NH   16
#define HD   64

typedef __attribute__((ext_vector_type(16))) _Float16 v16h;
typedef __attribute__((ext_vector_type(8)))  float    v8f;
typedef __attribute__((ext_vector_type(4)))  unsigned int v4u;
typedef __attribute__((ext_vector_type(8)))  int      v8i;
typedef __attribute__((ext_vector_type(4)))  int      v4i;

#if __has_builtin(__builtin_amdgcn_tensor_load_to_lds) && \
    __has_builtin(__builtin_amdgcn_s_wait_tensorcnt)
#define USE_TDM 1
#endif

// ---- WMMA wrapper: D = A(16x32 f16) * B(32x16 f16) + C(16x16 f32) ----------
__device__ __forceinline__ v8f wmma16(v16h a, v16h b, v8f c) {
  return __builtin_amdgcn_wmma_f32_16x16x32_f16(
      /*neg_a=*/false, a, /*neg_b=*/false, b,
      /*c_mod=*/(short)0, c, /*reuse_a=*/false, /*reuse_b=*/false);
}

// ---- A fragment (16x32, f16) from LDS tile stored row-major [M][K] ---------
// lanes 0-15 rows M=0..15 hold K 0..15, lanes 16-31 same
// rows hold K 16..31 (matches B fragment order); 2 f16 per VGPR.
__device__ __forceinline__ v16h ldsA16x32(const _Float16* p, int row0, int ld,
                                          int k0, int lane) {
  const int row  = row0 + (lane & 15);
  const int hofs = (lane >> 4) << 3;  // 0 or 8
  v16h a;
#pragma unroll
  for (int i = 0; i < 16; ++i) {
    const int v  = i >> 1;
    const int kk = (hofs << 1) + (v << 1) + (i & 1);
    a[i] = p[row * ld + k0 + kk];
  }
  return a;
}

// ---- B fragment (32x16, f16), tile stored [N][K] (e.g. weight rows) --------
// lanes 0-15: col N=lane, K=0..15 ; lanes 16-31: col N=lane-16, K=16..31
__device__ __forceinline__ v16h ldsB_nk(const _Float16* p, int n0, int ld,
                                        int k0, int lane) {
  const int n  = n0 + (lane & 15);
  const int kb = k0 + ((lane >> 4) << 4);
  v16h b;
#pragma unroll
  for (int i = 0; i < 16; ++i) b[i] = p[n * ld + kb + i];
  return b;
}

// ---- B fragment (32x16, f16), tile stored [K][N] (e.g. V tile) -------------
__device__ __forceinline__ v16h ldsB_kn(const _Float16* p, int k0, int ld,
                                        int n0, int lane) {
  const int n  = n0 + (lane & 15);
  const int kb = k0 + ((lane >> 4) << 4);
  v16h b;
#pragma unroll
  for (int i = 0; i < 16; ++i) b[i] = p[(kb + i) * ld + n];
  return b;
}

#ifdef USE_TDM
// ---- Tensor Data Mover: 2D f16 tile (tile_rows x 64) global -> LDS ---------
// Descriptor per CDNA5 ISA. data_size=2B,
// tile_dim0 = 64 elements (128 B rows). LDS pad feature reproduces the
// (HD+8) f16 row stride used by the compute: pad_interval code 4 = every
// 128 B, pad_amount code 3 = 16 B (4 DWORDs).
// This toolchain's builtin arity is 6: (v4u, v8i, v4i, v4i, v8i, i32 cpol);
// groups 2/3 (and the trailing group) are unused for 2D tiles -> zeros.
__device__ __forceinline__ void tdm_load_tile_f16(unsigned lds_off,
                                                  const _Float16* gptr,
                                                  unsigned tile_rows,
                                                  unsigned tensor_rows) {
  const unsigned long long ga = (unsigned long long)(size_t)gptr;
  v4u g0;
  g0[0] = 1u;                                     // count=1 user descriptor
  g0[1] = lds_off;                                // lds_addr (bytes)
  g0[2] = (unsigned)(ga & 0xffffffffu);           // global_addr[31:0]
  g0[3] = (unsigned)((ga >> 32) & 0x01ffffffu)    // global_addr[56:32]
          | (2u << 30);                           // type = 2 (image)
  v8i g1;
  g1[0] = (int)((1u << 16)      // data_size = 1 -> 2 bytes
                | (1u << 20)    // pad_enable
                | (4u << 22)    // pad_interval code 4 -> every 32 DWORDs
                | (3u << 25));  // pad_amount  code 3 -> 4 DWORDs (16 B)
  g1[1] = (int)(64u << 16);                       // tensor_dim0 = 64 (lo16)
  g1[2] = (int)((tensor_rows & 0xffffu) << 16);   // tensor_dim1 lo16
  g1[3] = (int)(((tensor_rows >> 16) & 0xffffu)   // tensor_dim1 hi16
                | (64u << 16));                   // tile_dim0 = 64
  g1[4] = (int)(tile_rows & 0xffffu);             // tile_dim1, tile_dim2=0
  g1[5] = 64;                                     // tensor_dim0_stride lo32
  g1[6] = 0;                                      // stride0 hi / stride1 lo
  g1[7] = 0;                                      // stride1 hi
  const v4i z4 = {};
  const v8i z8 = {};
  __builtin_amdgcn_tensor_load_to_lds(g0, g1, z4, z4, z8, 0);
}
#endif

// ===========================================================================
// Kernel 1: q/k/v projections + RoPE.
//   grid = (M/128, HID/64, 3)  block = 256 (8 waves, 4x2 wave grid)
//   Each 64-wide N tile is exactly one head (HD == 64).
// ===========================================================================
#define TM 128
#define TN 64
#define TK 32
#define LDT (TK + 8)

__global__ void __launch_bounds__(256)
qkv_rope_gemm(const float* __restrict__ x, const float* __restrict__ Wq,
              const float* __restrict__ Wk, const float* __restrict__ Wv,
              const float* __restrict__ cosp, const float* __restrict__ sinp,
              _Float16* __restrict__ qws, _Float16* __restrict__ kws,
              _Float16* __restrict__ vws) {
  __shared__ _Float16 xs[TM][LDT];
  __shared__ _Float16 wsh[TN][LDT];
  __shared__ float    cbuf[TM][TN + 2];

  const int tid  = threadIdx.x;
  const int lane = tid & 31;
  const int w    = tid >> 5;
  const int wm   = (w & 3) * 32;   // wave M offset within tile
  const int wn   = (w >> 2) * 32;  // wave N offset within tile
  const int ln   = lane & 15;
  const int half = lane >> 4;
  const int mb = blockIdx.x * TM;
  const int nb = blockIdx.y * TN;
  const int z  = blockIdx.z;

  const float* W = (z == 0) ? Wq : (z == 1) ? Wk : Wv;
  _Float16* outp = (z == 0) ? qws : (z == 1) ? kws : vws;

  v8f acc00 = {}, acc01 = {}, acc10 = {}, acc11 = {};

  for (int k0 = 0; k0 < HID; k0 += TK) {
#pragma unroll 4
    for (int idx = tid; idx < TM * TK; idx += 256) {
      const int r = idx >> 5, c = idx & 31;
      xs[r][c] = (_Float16)x[(size_t)(mb + r) * HID + k0 + c];
    }
#pragma unroll 2
    for (int idx = tid; idx < TN * TK; idx += 256) {
      const int r = idx >> 5, c = idx & 31;
      wsh[r][c] = (_Float16)W[(size_t)(nb + r) * HID + k0 + c];
    }
    // Prefetch next K-step tiles into cache while this step computes.
    if (k0 + TK < HID) {
      __builtin_prefetch(&x[(size_t)(mb + (tid >> 1)) * HID + k0 + TK +
                            ((tid & 1) << 4)], 0, 1);
      if (tid < 128)
        __builtin_prefetch(&W[(size_t)(nb + (tid >> 1)) * HID + k0 + TK +
                              ((tid & 1) << 4)], 0, 1);
    }
    __syncthreads();

    const v16h a0 = ldsA16x32(&xs[0][0], wm, LDT, 0, lane);
    const v16h a1 = ldsA16x32(&xs[0][0], wm + 16, LDT, 0, lane);
    const v16h b0 = ldsB_nk(&wsh[0][0], wn, LDT, 0, lane);
    const v16h b1 = ldsB_nk(&wsh[0][0], wn + 16, LDT, 0, lane);
    acc00 = wmma16(a0, b0, acc00);
    acc01 = wmma16(a0, b1, acc01);
    acc10 = wmma16(a1, b0, acc10);
    acc11 = wmma16(a1, b1, acc11);
    __syncthreads();
  }

  // Stage C fragments (lane = N col, VGPR r = row r + 8*half) to LDS.
#pragma unroll
  for (int r = 0; r < 8; ++r) {
    cbuf[wm + r + 8 * half][wn + ln]           = acc00[r];
    cbuf[wm + r + 8 * half][wn + 16 + ln]      = acc01[r];
    cbuf[wm + 16 + r + 8 * half][wn + ln]      = acc10[r];
    cbuf[wm + 16 + r + 8 * half][wn + 16 + ln] = acc11[r];
  }
  __syncthreads();

  const int h = blockIdx.y;  // one head per 64-wide N tile
  if (z == 2) {
    // V: plain f16 store to [B][NH][T][HD]
#pragma unroll 4
    for (int idx = tid; idx < TM * TN; idx += 256) {
      const int r = idx >> 6, c = idx & 63;
      const int m = mb + r, b = m >> 11, t = m & 2047;
      outp[((size_t)(b * NH + h) * TT + t) * HD + c] = (_Float16)cbuf[r][c];
    }
  } else {
    // Q/K: RoPE. cos/sin are [T][64] with duplicated halves.
#pragma unroll 2
    for (int idx = tid; idx < TM * 32; idx += 256) {
      const int r = idx >> 5, c = idx & 31;
      const int m = mb + r, b = m >> 11, t = m & 2047;
      const float cv = cosp[t * HD + c];
      const float sv = sinp[t * HD + c];
      const float x1 = cbuf[r][c];
      const float x2 = cbuf[r][c + 32];
      const size_t base = ((size_t)(b * NH + h) * TT + t) * HD;
      outp[base + c]      = (_Float16)(x1 * cv - x2 * sv);
      outp[base + c + 32] = (_Float16)(x2 * cv + x1 * sv);
    }
  }
}

// ===========================================================================
// Kernel 2: causal flash attention over f16 q/k/v ([BH][T][HD]).
//   grid = (T/128, B*NH)  block = 256; each wave owns 16 query rows.
//   Q/K/V tiles are moved into LDS by the Tensor Data Mover (one issuing
//   wave, s_wait_tensorcnt, then workgroup barrier publishes the tile).
// ===========================================================================
__global__ void __launch_bounds__(256)
flash_attn(const _Float16* __restrict__ qws, const _Float16* __restrict__ kws,
           const _Float16* __restrict__ vws, _Float16* __restrict__ attn) {
  __shared__ _Float16 qs[128][HD + 8];
  __shared__ _Float16 ks[32][HD + 8];
  __shared__ _Float16 vs[32][HD + 8];
  __shared__ _Float16 ps[8][16][40];  // per-wave P staging (C-layout -> A-layout)

  const int tid = threadIdx.x, lane = tid & 31, w = tid >> 5;
  const int ln = lane & 15, half = lane >> 4;
  const int qb = blockIdx.x * 128;
  const int bh = blockIdx.y;
  const size_t base = (size_t)bh * TT * HD;

#ifdef USE_TDM
  // Q tile via TDM (completion covered by the first in-loop tensorcnt wait:
  // TENSORcnt ops from one wave complete in order).
  if (w == 0) {
    tdm_load_tile_f16((unsigned)(size_t)&qs[0][0], qws + base + (size_t)qb * HD,
                      128u, (unsigned)(TT - qb));
  }
#else
#pragma unroll 4
  for (int idx = tid; idx < 128 * HD; idx += 256) {
    const int r = idx >> 6, c = idx & 63;
    qs[r][c] = qws[base + (size_t)(qb + r) * HD + c];
  }
#endif

  const int m0 = w * 16;  // wave's query rows within tile
  v8f acc[4];
#pragma unroll
  for (int nt = 0; nt < 4; ++nt) acc[nt] = (v8f){};
  float mrow[8], lrow[8];
#pragma unroll
  for (int r = 0; r < 8; ++r) { mrow[r] = -1e30f; lrow[r] = 0.0f; }

  for (int j0 = 0; j0 < qb + 128; j0 += 32) {
    __syncthreads();  // all waves done reading previous K/V tiles
#ifdef USE_TDM
    if (w == 0) {
      tdm_load_tile_f16((unsigned)(size_t)&ks[0][0],
                        kws + base + (size_t)j0 * HD, 32u, (unsigned)(TT - j0));
      tdm_load_tile_f16((unsigned)(size_t)&vs[0][0],
                        vws + base + (size_t)j0 * HD, 32u, (unsigned)(TT - j0));
      __builtin_amdgcn_s_wait_tensorcnt(0);
    }
#else
#pragma unroll 2
    for (int idx = tid; idx < 32 * HD; idx += 256) {
      const int r = idx >> 6, c = idx & 63;
      ks[r][c] = kws[base + (size_t)(j0 + r) * HD + c];
      vs[r][c] = vws[base + (size_t)(j0 + r) * HD + c];
    }
#endif
    __syncthreads();

    // S = Q (16x64) @ K^T (64x32): two N tiles, two K steps.
    v8f s0 = {}, s1 = {};
    {
      v16h aq = ldsA16x32(&qs[0][0], m0, HD + 8, 0, lane);
      v16h b0 = ldsB_nk(&ks[0][0], 0, HD + 8, 0, lane);
      v16h b1 = ldsB_nk(&ks[0][0], 16, HD + 8, 0, lane);
      s0 = wmma16(aq, b0, s0);
      s1 = wmma16(aq, b1, s1);
      aq = ldsA16x32(&qs[0][0], m0, HD + 8, 32, lane);
      b0 = ldsB_nk(&ks[0][0], 0, HD + 8, 32, lane);
      b1 = ldsB_nk(&ks[0][0], 16, HD + 8, 32, lane);
      s0 = wmma16(aq, b0, s0);
      s1 = wmma16(aq, b1, s1);
    }

    const float scale = 0.125f;  // 1/sqrt(64)
    const int key0 = j0 + ln, key1 = j0 + 16 + ln;
#pragma unroll
    for (int r = 0; r < 8; ++r) {
      const int qrow = qb + m0 + r + 8 * half;
      float v0 = s0[r] * scale, v1 = s1[r] * scale;
      if (key0 > qrow) v0 = -1e30f;  // causal mask
      if (key1 > qrow) v1 = -1e30f;
      float tmax = fmaxf(v0, v1);
      tmax = fmaxf(tmax, __shfl_xor(tmax, 1, 32));
      tmax = fmaxf(tmax, __shfl_xor(tmax, 2, 32));
      tmax = fmaxf(tmax, __shfl_xor(tmax, 4, 32));
      tmax = fmaxf(tmax, __shfl_xor(tmax, 8, 32));
      const float mnew = fmaxf(mrow[r], tmax);
      const float corr = __expf(mrow[r] - mnew);
      const float p0 = __expf(v0 - mnew);
      const float p1 = __expf(v1 - mnew);
      float psum = p0 + p1;
      psum += __shfl_xor(psum, 1, 32);
      psum += __shfl_xor(psum, 2, 32);
      psum += __shfl_xor(psum, 4, 32);
      psum += __shfl_xor(psum, 8, 32);
      lrow[r] = lrow[r] * corr + psum;
      mrow[r] = mnew;
#pragma unroll
      for (int nt = 0; nt < 4; ++nt) acc[nt][r] *= corr;
      ps[w][r + 8 * half][ln]      = (_Float16)p0;
      ps[w][r + 8 * half][16 + ln] = (_Float16)p1;
    }

    // O += P (16x32) @ V (32x64); private staging area -> no barrier needed.
    const v16h ap = ldsA16x32(&ps[w][0][0], 0, 40, 0, lane);
#pragma unroll
    for (int nt = 0; nt < 4; ++nt) {
      const v16h bv = ldsB_kn(&vs[0][0], 0, HD + 8, nt * 16, lane);
      acc[nt] = wmma16(ap, bv, acc[nt]);
    }
  }

  // Epilogue: normalize and store f16 attn as [B][T][HID] (head-interleaved).
  const int b = bh >> 4, h = bh & 15;
#pragma unroll
  for (int nt = 0; nt < 4; ++nt) {
#pragma unroll
    for (int r = 0; r < 8; ++r) {
      const int t = qb + m0 + r + 8 * half;
      const float o = acc[nt][r] / lrow[r];
      attn[((size_t)b * TT + t) * HID + h * HD + nt * 16 + ln] = (_Float16)o;
    }
  }
}

// ===========================================================================
// Kernel 3: out = attn (f16) @ Wo^T -> f32.  grid = (M/128, HID/64)
// ===========================================================================
__global__ void __launch_bounds__(256)
out_proj_gemm(const _Float16* __restrict__ attn, const float* __restrict__ Wo,
              float* __restrict__ out) {
  __shared__ _Float16 as_[TM][LDT];
  __shared__ _Float16 wsh[TN][LDT];

  const int tid  = threadIdx.x;
  const int lane = tid & 31;
  const int w    = tid >> 5;
  const int wm   = (w & 3) * 32;
  const int wn   = (w >> 2) * 32;
  const int ln   = lane & 15;
  const int half = lane >> 4;
  const int mb = blockIdx.x * TM;
  const int nb = blockIdx.y * TN;

  v8f acc00 = {}, acc01 = {}, acc10 = {}, acc11 = {};

  for (int k0 = 0; k0 < HID; k0 += TK) {
#pragma unroll 4
    for (int idx = tid; idx < TM * TK; idx += 256) {
      const int r = idx >> 5, c = idx & 31;
      as_[r][c] = attn[(size_t)(mb + r) * HID + k0 + c];
    }
#pragma unroll 2
    for (int idx = tid; idx < TN * TK; idx += 256) {
      const int r = idx >> 5, c = idx & 31;
      wsh[r][c] = (_Float16)Wo[(size_t)(nb + r) * HID + k0 + c];
    }
    if (k0 + TK < HID) {
      __builtin_prefetch(&attn[(size_t)(mb + (tid >> 1)) * HID + k0 + TK +
                               ((tid & 1) << 4)], 0, 1);
      if (tid < 128)
        __builtin_prefetch(&Wo[(size_t)(nb + (tid >> 1)) * HID + k0 + TK +
                               ((tid & 1) << 4)], 0, 1);
    }
    __syncthreads();

    const v16h a0 = ldsA16x32(&as_[0][0], wm, LDT, 0, lane);
    const v16h a1 = ldsA16x32(&as_[0][0], wm + 16, LDT, 0, lane);
    const v16h b0 = ldsB_nk(&wsh[0][0], wn, LDT, 0, lane);
    const v16h b1 = ldsB_nk(&wsh[0][0], wn + 16, LDT, 0, lane);
    acc00 = wmma16(a0, b0, acc00);
    acc01 = wmma16(a0, b1, acc01);
    acc10 = wmma16(a1, b0, acc10);
    acc11 = wmma16(a1, b1, acc11);
    __syncthreads();
  }

  // Direct f32 store from C fragments.
#pragma unroll
  for (int r = 0; r < 8; ++r) {
    const int row0 = mb + wm + r + 8 * half;
    const int row1 = mb + wm + 16 + r + 8 * half;
    out[(size_t)row0 * HID + nb + wn + ln]      = acc00[r];
    out[(size_t)row0 * HID + nb + wn + 16 + ln] = acc01[r];
    out[(size_t)row1 * HID + nb + wn + ln]      = acc10[r];
    out[(size_t)row1 * HID + nb + wn + 16 + ln] = acc11[r];
  }
}

// ===========================================================================
extern "C" void kernel_launch(void* const* d_in, const int* in_sizes, int n_in,
                              void* d_out, int out_size, void* d_ws,
                              size_t ws_size, hipStream_t stream) {
  const float* x    = (const float*)d_in[0];
  const float* Wq   = (const float*)d_in[1];
  const float* Wk   = (const float*)d_in[2];
  const float* Wv   = (const float*)d_in[3];
  const float* Wo   = (const float*)d_in[4];
  const float* cosp = (const float*)d_in[5];
  const float* sinp = (const float*)d_in[6];
  float* out = (float*)d_out;

  const size_t per = (size_t)BB * NH * TT * HD;  // 4 Mi f16 elements
  _Float16* qws  = (_Float16*)d_ws;
  _Float16* kws  = qws + per;
  _Float16* vws  = kws + per;
  _Float16* attn = vws + per;

  // 1) Q/K/V projections + RoPE
  dim3 g1((BB * TT) / TM, HID / TN, 3);
  qkv_rope_gemm<<<g1, 256, 0, stream>>>(x, Wq, Wk, Wv, cosp, sinp, qws, kws,
                                        vws);

  // 2) causal flash attention
  dim3 g2(TT / 128, BB * NH);
  flash_attn<<<g2, 256, 0, stream>>>(qws, kws, vws, attn);

  // 3) output projection
  dim3 g3((BB * TT) / TM, HID / TN);
  out_proj_gemm<<<g3, 256, 0, stream>>>(attn, Wo, out);
}
